// BidirectionalPropagation_49297634624013
// MI455X (gfx1250) — matrix operand order repaired and checked by reading, not compiled, not run on hardware
//
#include <hip/hip_runtime.h>
#include <hip/hip_bf16.h>
#include <stdint.h>

// ---------------------------------------------------------------------------
// BidirectionalPropagation (BasicVSR++ style) for MI455X / gfx1250.
//
// Every conv (3x3 offset net, deformable conv einsum, backbone, 1x1 fusion)
// is an implicit GEMM:  out[M,4096] = A_f16[M,K] x Bt_f16[4096,K]^T
// (+bias, leaky-relu, residual) on v_wmma_f32_16x16x32_f16 (wave32).
// Producers write the im2col operand TRANSPOSED (Bt[p][k]) so both GEMM
// operands are row-strips -> staged with global_load_async_to_lds_b128 and
// double-buffered (compute overlaps the async copies; s_wait_asynccnt 0 +
// one barrier per K-step).
//
// Input index assumption (top-level dict order, params flattened sorted):
//   0:x 1:flows_backward 2:flows_forward
//   3..16  backward_: bb_b1,bb_b2,bb_w1,bb_w2,da_b,da_w,
//                     off_b1..off_b4,off_w1..off_w4
//   17..30 forward_ : same order (bb_w1 has Cin=384)
//   31:fusion_b 32:fusion_w
// ---------------------------------------------------------------------------

typedef __attribute__((ext_vector_type(16))) _Float16 v16h;
typedef __attribute__((ext_vector_type(8)))  float    v8f;
typedef int v4i __attribute__((vector_size(16)));

#define HH 64
#define WW 64
#define PP 4096
#define CCH 128
#define CHW (128 * 4096)

#if defined(__has_builtin)
#if __has_builtin(__builtin_amdgcn_global_load_async_to_lds_b128) && \
    __has_builtin(__builtin_amdgcn_s_wait_asynccnt)
#define USE_ASYNC_LDS 1
#else
#define USE_ASYNC_LDS 0
#endif
#else
#define USE_ASYNC_LDS 0
#endif

__device__ __forceinline__ float lrelu01(float v) { return v > 0.f ? v : 0.1f * v; }

__device__ __forceinline__ float bilin0(const float* __restrict__ img, float ys, float xs) {
  float y0 = floorf(ys), x0 = floorf(xs);
  float wy = ys - y0, wx = xs - x0;
  int iy = (int)y0, ix = (int)x0;
  float acc = 0.f;
#pragma unroll
  for (int dy = 0; dy < 2; ++dy) {
#pragma unroll
    for (int dx = 0; dx < 2; ++dx) {
      int yy = iy + dy, xx = ix + dx;
      if ((unsigned)yy < (unsigned)HH && (unsigned)xx < (unsigned)WW) {
        float wgt = (dy ? wy : 1.f - wy) * (dx ? wx : 1.f - wx);
        acc += img[yy * WW + xx] * wgt;
      }
    }
  }
  return acc;
}

// ---------------------------------------------------------------------------
// weights: OIHW f32 -> f16 [Mpad][Kpad] row-major (K = Cin*kh*kw)
// ---------------------------------------------------------------------------
__global__ void k_wcvt(const float* __restrict__ w, _Float16* __restrict__ A,
                       int M, int K, int Kp, int Mp) {
  int idx = blockIdx.x * 256 + threadIdx.x;
  if (idx >= Mp * Kp) return;
  int m = idx / Kp;
  int k = idx - m * Kp;
  float v = (m < M && k < K) ? w[(size_t)m * K + k] : 0.f;
  A[idx] = (_Float16)v;
}

__global__ void k_fill0(float* __restrict__ p, int n) {
  int idx = blockIdx.x * 256 + threadIdx.x;
  if (idx < n) p[idx] = 0.f;
}

// ---------------------------------------------------------------------------
// im2col (3x3 pad1 or 1x1) from up to 5 concatenated NCHW f32 sources.
// Writes TRANSPOSED f16 operand: Bt[p*Kp + k]  (p-major -> coalesced stores).
// ---------------------------------------------------------------------------
struct Src5 {
  const float* p0; const float* p1; const float* p2; const float* p3; const float* p4;
  int c0, c1, c2, c3, c4;
};

__global__ void k_im2col(Src5 s, _Float16* __restrict__ Bt, int Cin, int Kp, int ks3) {
  int idx = blockIdx.x * 256 + threadIdx.x;
  if (idx >= Kp * PP) return;
  int pp = idx / Kp;
  int k = idx - pp * Kp;
  float v = 0.f;
  int Ktot = ks3 ? Cin * 9 : Cin;
  if (k < Ktot) {
    int ci, y, x;
    if (ks3) {
      ci = k / 9;
      int t = k - ci * 9;
      y = (pp >> 6) + t / 3 - 1;
      x = (pp & 63) + (t % 3) - 1;
    } else {
      ci = k; y = pp >> 6; x = pp & 63;
    }
    if ((unsigned)y < (unsigned)HH && (unsigned)x < (unsigned)WW) {
      const float* src = nullptr;
      int cl = ci;
      if (cl < s.c0) { src = s.p0; }
      else { cl -= s.c0;
        if (cl < s.c1) { src = s.p1; }
        else { cl -= s.c1;
          if (cl < s.c2) { src = s.p2; }
          else { cl -= s.c2;
            if (cl < s.c3) { src = s.p3; }
            else { cl -= s.c3; src = s.p4; }
          }
        }
      }
      if (src) v = src[(size_t)cl * PP + y * WW + x];
    }
  }
  Bt[idx] = (_Float16)v;
}

// ---------------------------------------------------------------------------
// deformable-conv im2col (K = 2C*9 = 2304), transposed output Bt[p*2304+k].
// Fused: 10*tanh(offset)+flow, sigmoid(mask), bilinear gather.
//   offset channel = g*18 + t*2 + d (d=0 -> y += flow[1], d=1 -> x += flow[0])
//   mask   channel = 288 + g*9 + t ; groups g<8 use flow_1, g>=8 use flow_2.
// ---------------------------------------------------------------------------
__global__ void k_deform_b(const float* __restrict__ o,
                           const float* __restrict__ fl1,
                           const float* __restrict__ fl2,
                           const float* __restrict__ x0,
                           const float* __restrict__ x1,
                           _Float16* __restrict__ Bt) {
  int idx = blockIdx.x * 256 + threadIdx.x;
  if (idx >= 2304 * PP) return;
  int pp = idx / 2304;
  int k = idx - pp * 2304;
  int ci = k / 9;
  int t = k - ci * 9;
  int g = ci >> 4;                 // Cg = 16
  const float* fl = (g < 8) ? fl1 : fl2;
  int och = g * 18 + t * 2;
  float oy = 10.f * tanhf(o[(size_t)och * PP + pp]) + fl[PP + pp];
  float ox = 10.f * tanhf(o[(size_t)(och + 1) * PP + pp]) + fl[pp];
  float mk = 1.f / (1.f + expf(-o[(size_t)(288 + g * 9 + t) * PP + pp]));
  float ys = (float)(pp >> 6) + (float)(t / 3 - 1) + oy;
  float xs = (float)(pp & 63) + (float)(t % 3 - 1) + ox;
  const float* src = (ci < CCH) ? (x0 + (size_t)ci * PP) : (x1 + (size_t)(ci - CCH) * PP);
  Bt[idx] = (_Float16)(bilin0(src, ys, xs) * mk);
}

// ---------------------------------------------------------------------------
// flow warp (bilinear, zero pad); optional elementwise add of `add`
// ---------------------------------------------------------------------------
__global__ void k_warp(const float* __restrict__ src, int ch,
                       const float* __restrict__ flow,
                       const float* __restrict__ add,
                       float* __restrict__ out) {
  int idx = blockIdx.x * 256 + threadIdx.x;
  if (idx >= ch * PP) return;
  int c = idx >> 12;
  int pp = idx & (PP - 1);
  float ys = (float)(pp >> 6) + flow[PP + pp];
  float xs = (float)(pp & 63) + flow[pp];
  float v = bilin0(src + (size_t)c * PP, ys, xs);
  if (add) v += add[idx];
  out[idx] = v;
}

// ---------------------------------------------------------------------------
// WMMA GEMM:  out[M,4096] = A[Mpad,Kp] x Bt[4096,Kp]^T  (+bias, act, +resid)
// 256 threads = 8 waves; tile BM=128 x BN=128; K-step 32; double-buffered LDS.
// Wave (mw 0..3, nw 0..1): 2 A-frags x 4 B-frags -> 8 wmma / K-step.
// Both operand tiles are row-strips -> staged via async b128 copies.
// Fragment layouts per CDNA5 ISA 7.12.2.
// ---------------------------------------------------------------------------
#define BM 128
#define BN 128
#define BK 32
#define LDF 40   // halves per LDS row: 80B (16B-aligned segments for b128 async)

__device__ __forceinline__ void stage_row_async(const _Float16* __restrict__ g,
                                                _Float16* l) {
#if USE_ASYNC_LDS
  __builtin_amdgcn_global_load_async_to_lds_b128(
      (__attribute__((address_space(1))) v4i*)(uintptr_t)g,
      (__attribute__((address_space(3))) v4i*)(uintptr_t)l,
      0, 0);
#else
  *(v4i*)l = *(const v4i*)g;
#endif
}

__global__ __launch_bounds__(256) void k_gemm(const _Float16* __restrict__ A,
                                              const _Float16* __restrict__ Bt,
                                              const float* __restrict__ bias,
                                              const float* __restrict__ resid,
                                              float* __restrict__ out,
                                              int M, int Kp, int act) {
  __shared__ _Float16 lsA[2][BM * LDF];
  __shared__ _Float16 lsB[2][BN * LDF];

  const int tid = threadIdx.x;
  const int lane = tid & 31;
  const int wv = tid >> 5;
  const int mw = wv & 3;        // 0..3 -> 32-row M strip
  const int nw = wv >> 2;       // 0..1 -> 64-col N half
  const int pBase = blockIdx.x * BN;
  const int mBase = blockIdx.y * BM;

  v8f acc[2][4] = {};

  auto stage = [&](int buf, int k0) {
    // A tile: 128 rows x 32 halves, 4x b128 segments per row
#pragma unroll
    for (int it = 0; it < 2; ++it) {
      int e = tid + it * 256;
      int row = e >> 2, seg = e & 3;
      stage_row_async(&A[(size_t)(mBase + row) * Kp + k0 + seg * 8],
                      &lsA[buf][row * LDF + seg * 8]);
    }
    // B tile: 128 rows (pixels) x 32 halves
#pragma unroll
    for (int it = 0; it < 2; ++it) {
      int e = tid + it * 256;
      int row = e >> 2, seg = e & 3;
      stage_row_async(&Bt[(size_t)(pBase + row) * Kp + k0 + seg * 8],
                      &lsB[buf][row * LDF + seg * 8]);
    }
  };

  const int nIter = Kp / BK;
  stage(0, 0);
#if USE_ASYNC_LDS
  __builtin_amdgcn_s_wait_asynccnt(0);
#endif
  __syncthreads();

  for (int ii = 0; ii < nIter; ++ii) {
    const int buf = ii & 1;
    if (ii + 1 < nIter) stage(buf ^ 1, (ii + 1) * BK);   // overlap with compute

    // A fragments (16x32 f16): lane L -> M=(L&15); half=L>>4;
    // VGPR v holds K=(v/4)*16 + half*8 + (v%4)*2 (+0,+1)
    union U { v16h h; uint32_t u[8]; };
    U af[2];
    const int ahalf = lane >> 4;
#pragma unroll
    for (int i = 0; i < 2; ++i) {
      const int am = mw * 32 + i * 16 + (lane & 15);
#pragma unroll
      for (int v = 0; v < 8; ++v) {
        int kk = ((v >> 2) << 4) + (ahalf << 3) + ((v & 3) << 1);
        af[i].u[v] = *reinterpret_cast<const uint32_t*>(&lsA[buf][am * LDF + kk]);
      }
    }
#pragma unroll
    for (int j = 0; j < 4; ++j) {
      // B fragment: lane L -> N=(L&15); VGPR v holds K=(v/4)*16 + (L>>4)*8 + (v%4)*2 (+0,+1)
      U bf;
      const int bn = nw * 64 + j * 16 + (lane & 15);
      const int bh = lane >> 4;
#pragma unroll
      for (int v = 0; v < 8; ++v) {
        int kk = ((v >> 2) << 4) + (bh << 3) + ((v & 3) << 1);
        bf.u[v] = *reinterpret_cast<const uint32_t*>(&lsB[buf][bn * LDF + kk]);
      }
      acc[0][j] = __builtin_amdgcn_wmma_f32_16x16x32_f16(
          false, af[0].h, false, bf.h, (short)0, acc[0][j], false, false);
      acc[1][j] = __builtin_amdgcn_wmma_f32_16x16x32_f16(
          false, af[1].h, false, bf.h, (short)0, acc[1][j], false, false);
    }
#if USE_ASYNC_LDS
    __builtin_amdgcn_s_wait_asynccnt(0);   // next buffer's copies done
#endif
    __syncthreads();
  }

  // epilogue: C/D layout lane L -> N=L&15, VGPR r -> M=(L>>4)*8 + r
  const int nlane = lane & 15;
  const int chalf = lane >> 4;
#pragma unroll
  for (int i = 0; i < 2; ++i) {
#pragma unroll
    for (int j = 0; j < 4; ++j) {
      int n = pBase + nw * 64 + j * 16 + nlane;
#pragma unroll
      for (int r = 0; r < 8; ++r) {
        int m = mBase + mw * 32 + i * 16 + chalf * 8 + r;
        if (m < M) {
          float v = acc[i][j][r] + bias[m];
          if (act) v = lrelu01(v);
          size_t off = (size_t)m * PP + n;
          if (resid) v += resid[off];
          out[off] = v;
        }
      }
    }
  }
}

// ---------------------------------------------------------------------------
// host orchestration
// ---------------------------------------------------------------------------
static inline int cdiv(int a, int b) { return (a + b - 1) / b; }

extern "C" void kernel_launch(void* const* d_in, const int* in_sizes, int n_in,
                              void* d_out, int out_size, void* d_ws, size_t ws_size,
                              hipStream_t stream) {
  (void)in_sizes; (void)out_size; (void)ws_size;
  if (n_in < 33) return;

  const float* x   = (const float*)d_in[0];   // (10,128,64,64)
  const float* fbw = (const float*)d_in[1];   // (9,2,64,64)
  const float* ffw = (const float*)d_in[2];
  const float* fus_b = (const float*)d_in[31];
  const float* fus_w = (const float*)d_in[32];
  float* outp = (float*)d_out;

  char* base = (char*)d_ws;
  size_t used = 0;
  auto alloc = [&](size_t bytes) -> void* {
    size_t a = (used + 255) & ~(size_t)255;
    void* r = base + a;
    used = a + bytes;
    return r;
  };

  float* Z  = (float*)alloc(CHW * 4);
  float* FB[10];
  for (int i = 0; i < 10; ++i) FB[i] = (float*)alloc(CHW * 4);
  float* T  = (float*)alloc(CHW * 4);
  float* C1 = (float*)alloc(CHW * 4);
  float* C2 = (float*)alloc(CHW * 4);
  float* oA = (float*)alloc(CHW * 4);
  float* oB = (float*)alloc(CHW * 4);
  float* o4 = (float*)alloc((size_t)432 * PP * 4);
  float* fn2buf = (float*)alloc(2 * PP * 4);
  float* FR[3];
  for (int i = 0; i < 3; ++i) FR[i] = (float*)alloc(CHW * 4);
  _Float16* Bbuf = (_Float16*)alloc((size_t)3520 * PP * 2);   // Bt[4096][Kp]

  auto fill = [&](float* p, int n) {
    k_fill0<<<cdiv(n, 256), 256, 0, stream>>>(p, n);
  };
  auto wcvt = [&](const float* w, int M, int K, int Kp, int Mp) -> _Float16* {
    _Float16* A = (_Float16*)alloc((size_t)Mp * Kp * 2);
    int n = Mp * Kp;
    k_wcvt<<<cdiv(n, 256), 256, 0, stream>>>(w, A, M, K, Kp, Mp);
    return A;
  };
  auto im2 = [&](const float* a, int ca, const float* b, int cb,
                 const float* c, int cc, const float* d, int cd,
                 const float* e, int ce, int Cin, int Kp, int ks3) {
    Src5 s{a, b, c, d, e, ca, cb, cc, cd, ce};
    int n = Kp * PP;
    k_im2col<<<cdiv(n, 256), 256, 0, stream>>>(s, Bbuf, Cin, Kp, ks3);
  };
  auto gemm = [&](const _Float16* A, const float* bias, const float* resid,
                  float* out, int M, int Mp, int Kp, int act) {
    dim3 g(PP / BN, Mp / BM);
    k_gemm<<<g, 256, 0, stream>>>(A, Bbuf, bias, resid, out, M, Kp, act);
  };

  // ---- convert all weights (every call; deterministic) ----
  struct Mod {
    const float *bb_b1, *bb_b2, *da_b, *off_b1, *off_b2, *off_b3, *off_b4;
    _Float16 *A_bb1, *A_bb2, *A_da, *A_o1, *A_o2, *A_o3, *A_o4;
    int K_bb1;
  } mod[2];
  for (int dir = 0; dir < 2; ++dir) {
    int pb = dir == 0 ? 3 : 17;      // backward_, forward_
    Mod& m = mod[dir];
    m.bb_b1  = (const float*)d_in[pb + 0];
    m.bb_b2  = (const float*)d_in[pb + 1];
    const float* bb_w1 = (const float*)d_in[pb + 2];
    const float* bb_w2 = (const float*)d_in[pb + 3];
    m.da_b   = (const float*)d_in[pb + 4];
    const float* da_w  = (const float*)d_in[pb + 5];
    m.off_b1 = (const float*)d_in[pb + 6];
    m.off_b2 = (const float*)d_in[pb + 7];
    m.off_b3 = (const float*)d_in[pb + 8];
    m.off_b4 = (const float*)d_in[pb + 9];
    const float* off_w1 = (const float*)d_in[pb + 10];
    const float* off_w2 = (const float*)d_in[pb + 11];
    const float* off_w3 = (const float*)d_in[pb + 12];
    const float* off_w4 = (const float*)d_in[pb + 13];
    m.K_bb1 = dir == 0 ? 2304 : 3456;            // (2C|3C)*9
    m.A_bb1 = wcvt(bb_w1, 128, m.K_bb1, m.K_bb1, 128);
    m.A_bb2 = wcvt(bb_w2, 128, 1152, 1152, 128);
    m.A_da  = wcvt(da_w,  128, 2304, 2304, 128);
    m.A_o1  = wcvt(off_w1, 128, 3492, 3520, 128); // (3C+4)*9 padded to /32
    m.A_o2  = wcvt(off_w2, 128, 1152, 1152, 128);
    m.A_o3  = wcvt(off_w3, 128, 1152, 1152, 128);
    m.A_o4  = wcvt(off_w4, 432, 1152, 1152, 512); // M padded to /128
  }
  _Float16* A_fus = wcvt(fus_w, 128, 256, 256, 128);

  fill(Z, CHW);

  // ---- propagation ----
  for (int dir = 0; dir < 2; ++dir) {
    const bool back = (dir == 0);
    const Mod& m = mod[dir];
    const float* flows = back ? fbw : ffw;
    float* prev = nullptr;
    float* prev2 = nullptr;

    for (int i = 0; i < 10; ++i) {
      int idx = back ? 9 - i : i;
      const float* feat_cur = x + (size_t)idx * CHW;
      const float* fp  = (i == 0) ? Z : prev;        // feat_prop (in)
      const float* fn2 = (i >= 2) ? prev2 : Z;       // feat_n2
      const float* propOut;

      if (i > 0) {
        const float* flow1 = flows + (size_t)(i - 1) * 2 * PP;
        const float* flow2;
        if (i > 1) {
          // flow_n2 = flow_n1 + warp(flows[i-2], flow_n1)
          k_warp<<<cdiv(2 * PP, 256), 256, 0, stream>>>(
              flows + (size_t)(i - 2) * 2 * PP, 2, flow1, flow1, fn2buf);
          flow2 = fn2buf;
          k_warp<<<cdiv(CHW, 256), 256, 0, stream>>>(fn2, 128, flow2, nullptr, C2);
        } else {
          fill(fn2buf, 2 * PP);
          flow2 = fn2buf;
          fill(C2, CHW);
        }
        // cond_n1 = warp(feat_prop, flow_n1)
        k_warp<<<cdiv(CHW, 256), 256, 0, stream>>>(fp, 128, flow1, nullptr, C1);

        // offset network: ef = [cond_n1, feat_cur, cond_n2, flow1, flow2]
        im2(C1, 128, feat_cur, 128, C2, 128, flow1, 2, flow2, 2, 388, 3520, 1);
        gemm(m.A_o1, m.off_b1, nullptr, oA, 128, 128, 3520, 1);
        im2(oA, 128, nullptr, 0, nullptr, 0, nullptr, 0, nullptr, 0, 128, 1152, 1);
        gemm(m.A_o2, m.off_b2, nullptr, oB, 128, 128, 1152, 1);
        im2(oB, 128, nullptr, 0, nullptr, 0, nullptr, 0, nullptr, 0, 128, 1152, 1);
        gemm(m.A_o3, m.off_b3, nullptr, oA, 128, 128, 1152, 1);
        im2(oA, 128, nullptr, 0, nullptr, 0, nullptr, 0, nullptr, 0, 128, 1152, 1);
        gemm(m.A_o4, m.off_b4, nullptr, o4, 432, 512, 1152, 0);

        // modulated deformable conv as GEMM (K = 2C*9 = 2304)
        {
          int n = 2304 * PP;
          k_deform_b<<<cdiv(n, 256), 256, 0, stream>>>(o4, flow1, flow2, fp, fn2, Bbuf);
        }
        gemm(m.A_da, m.da_b, nullptr, T, 128, 128, 2304, 0);
        propOut = T;
      } else {
        propOut = Z;
      }

      // backbone: feat_prop += conv2(lrelu(conv1(concat)))
      if (back) {
        im2(feat_cur, 128, propOut, 128, nullptr, 0, nullptr, 0, nullptr, 0,
            256, 2304, 1);
      } else {
        im2(feat_cur, 128, FB[idx], 128, propOut, 128, nullptr, 0, nullptr, 0,
            384, 3456, 1);
      }
      gemm(m.A_bb1, m.bb_b1, nullptr, oA, 128, 128, m.K_bb1, 1);
      im2(oA, 128, nullptr, 0, nullptr, 0, nullptr, 0, nullptr, 0, 128, 1152, 1);
      float* res = back ? FB[idx] : FR[i % 3];
      gemm(m.A_bb2, m.bb_b2, propOut, res, 128, 128, 1152, 0);

      if (!back) {
        // fusion (1x1) + global residual, straight into d_out
        im2(FB[idx], 128, res, 128, nullptr, 0, nullptr, 0, nullptr, 0,
            256, 256, 0);
        gemm(A_fus, fus_b, x + (size_t)idx * CHW, outp + (size_t)idx * CHW,
             128, 128, 256, 0);
      }
      prev2 = prev;
      prev = res;
    }
  }
}
